// CaseLevelContext_31078383354738
// MI455X (gfx1250) — matrix operand order repaired and checked by reading, not compiled, not run on hardware
//
#include <hip/hip_runtime.h>

typedef __attribute__((ext_vector_type(16))) _Float16 v16h;
typedef __attribute__((ext_vector_type(8)))  float    v8f;
typedef __attribute__((ext_vector_type(4)))  unsigned int v4u;
typedef int v4i_vs __attribute__((vector_size(16)));   // matches builtin param type

#if __has_builtin(__builtin_amdgcn_global_load_async_to_lds_b128)
#define HAS_ASYNC_LDS 1
#else
#define HAS_ASYNC_LDS 0
#endif

namespace {
constexpr int Bb = 64, Ss = 512, Dd = 400, Hh = 8, DH = 50, ATd = 400;
constexpr int KP  = 416;      // K padded (400 -> 13 * 32)
constexpr int DHP = 64;       // head dim padded (50 -> 2 * 32)
constexpr int MROWS = Bb * Ss;                 // 32768
constexpr int NCLS = 598, NPAD = 640, NQKV = 448;
constexpr int NCHUNK_QKV = 7;                  // ceil(448/64)
constexpr int NCHUNK_CLS = 10;                 // ceil(640/64)
constexpr float NEG = 1.0e7f;
}

union FragAB { v16h h; v4u u[2]; };

__device__ __forceinline__ v4u ldg4(const _Float16* p) { return *(const v4u*)p; }

__device__ __forceinline__ v8f wmma16(const FragAB& a, const FragAB& b, v8f c) {
  return __builtin_amdgcn_wmma_f32_16x16x32_f16(false, a.h, false, b.h,
                                                (short)0, c, false, false);
}

// 16-byte global -> LDS copy: async (ASYNCcnt) when available, else via VGPRs.
__device__ __forceinline__ void copy16_g2l(const void* g, void* l) {
#if HAS_ASYNC_LDS
  __builtin_amdgcn_global_load_async_to_lds_b128(
      (__attribute__((address_space(1))) v4i_vs*)g,
      (__attribute__((address_space(3))) v4i_vs*)l, 0, 0);
#else
  *(v4u*)l = *(const v4u*)g;
#endif
}

__device__ __forceinline__ void copy_wait() {
#if HAS_ASYNC_LDS
#if __has_builtin(__builtin_amdgcn_s_wait_asynccnt)
  __builtin_amdgcn_s_wait_asynccnt(0);
#else
  asm volatile("s_wait_asynccnt 0x0" ::: "memory");
#endif
#endif
  __syncthreads();
}

// ---------------- prep: f32 -> f16 with K padding ----------------
__global__ void k_cvt_x(const float* __restrict__ X, _Float16* __restrict__ Xh) {
  long i = (long)blockIdx.x * blockDim.x + threadIdx.x;
  if (i >= (long)MROWS * KP) return;
  int row = (int)(i / KP), col = (int)(i % KP);
  Xh[i] = (col < Dd) ? (_Float16)X[(long)row * Dd + col] : (_Float16)0.0f;
}

__global__ void k_cvt_w(const float* __restrict__ Wq, const float* __restrict__ Wk,
                        const float* __restrict__ Wv,
                        const float* __restrict__ Wc0, const float* __restrict__ Wc1,
                        const float* __restrict__ Wc2, const float* __restrict__ Wc3,
                        const float* __restrict__ bc0, const float* __restrict__ bc1,
                        const float* __restrict__ bc2, const float* __restrict__ bc3,
                        _Float16* __restrict__ Wh,     // [3][NQKV][KP]
                        _Float16* __restrict__ Wcat,   // [NPAD][KP]
                        float* __restrict__ bcat) {    // [NPAD]
  const long total_wh = 3L * NQKV * KP;
  const long total_wc = (long)NPAD * KP;
  long i = (long)blockIdx.x * blockDim.x + threadIdx.x;
  if (i < total_wh) {
    int p = (int)(i / ((long)NQKV * KP));
    long rem = i % ((long)NQKV * KP);
    int n = (int)(rem / KP), k = (int)(rem % KP);
    const float* W = (p == 0) ? Wq : ((p == 1) ? Wk : Wv);
    Wh[i] = (n < ATd && k < Dd) ? (_Float16)W[(long)n * Dd + k] : (_Float16)0.0f;
  } else if (i < total_wh + total_wc) {
    long j = i - total_wh;
    int n = (int)(j / KP), k = (int)(j % KP);
    float v = 0.0f;
    if (k < ATd) {
      if      (n < 25)   v = Wc0[(long)n * ATd + k];
      else if (n < 29)   v = Wc1[(long)(n - 25) * ATd + k];
      else if (n < 591)  v = Wc2[(long)(n - 29) * ATd + k];
      else if (n < NCLS) v = Wc3[(long)(n - 591) * ATd + k];
    }
    Wcat[j] = (_Float16)v;
  } else if (i < total_wh + total_wc + NPAD) {
    int n = (int)(i - total_wh - total_wc);
    float v = 0.0f;
    if      (n < 25)   v = bc0[n];
    else if (n < 29)   v = bc1[n - 25];
    else if (n < 591)  v = bc2[n - 29];
    else if (n < NCLS) v = bc3[n - 591];
    bcat[n] = v;
  }
}

// ---------------- QKV projection GEMM + ELU ----------------
// Y = elu(X @ W^T + b); one wave computes a 32(M) x 64(N) tile (8 WMMAs / K-step).
__global__ __launch_bounds__(32)
void k_qkv(const _Float16* __restrict__ Xh, const _Float16* __restrict__ Wh,
           const float* __restrict__ bq, const float* __restrict__ bk,
           const float* __restrict__ bv,
           _Float16* __restrict__ qh,   // [B,H,S,DHP]
           _Float16* __restrict__ kh,   // [B,H,S,DHP]
           _Float16* __restrict__ vT) { // [B,H,DHP,S]
  const int lane = threadIdx.x & 31;
  const int hi = lane >> 4, lo = lane & 15;
  const int mt = blockIdx.x / NCHUNK_QKV;
  const int nc = blockIdx.x % NCHUNK_QKV;
  const int p  = blockIdx.y;
  const int m0 = mt * 32, n0 = nc * 64;
  const _Float16* Wp = Wh + (long)p * NQKV * KP;
  const float* bias = (p == 0) ? bq : ((p == 1) ? bk : bv);

  v8f acc[8];
#pragma unroll
  for (int t = 0; t < 8; ++t) acc[t] = v8f{};

  const _Float16* arow0 = Xh + (long)(m0 + lo) * KP;
  const _Float16* arow1 = Xh + (long)(m0 + 16 + lo) * KP;
  for (int k0 = 0; k0 < KP; k0 += 32) {
    FragAB a0, a1;
    a0.u[0] = ldg4(arow0 + k0 + hi * 8);
    a0.u[1] = ldg4(arow0 + k0 + 16 + hi * 8);
    a1.u[0] = ldg4(arow1 + k0 + hi * 8);
    a1.u[1] = ldg4(arow1 + k0 + 16 + hi * 8);
    if (k0 + 32 < KP) {   // prefetch next K-step of the streamed A operand
      __builtin_prefetch(arow0 + k0 + 32 + hi * 8, 0, 0);
      __builtin_prefetch(arow1 + k0 + 32 + hi * 8, 0, 0);
    }
#pragma unroll
    for (int t = 0; t < 4; ++t) {
      FragAB bf;
      const _Float16* bp = Wp + (long)(n0 + t * 16 + lo) * KP + k0 + hi * 8;
      bf.u[0] = ldg4(bp);
      bf.u[1] = ldg4(bp + 16);
      acc[t]     = wmma16(a0, bf, acc[t]);
      acc[4 + t] = wmma16(a1, bf, acc[4 + t]);
    }
  }

#pragma unroll
  for (int t = 0; t < 4; ++t) {
    int n = n0 + t * 16 + lo;
    if (n >= ATd) continue;
    float bval = bias[n];
    int h = n / DH, dh = n % DH;
#pragma unroll
    for (int half = 0; half < 2; ++half) {
#pragma unroll
      for (int r = 0; r < 8; ++r) {
        int row = m0 + half * 16 + hi * 8 + r;
        float x = acc[half * 4 + t][r] + bval;
        float y = (x > 0.0f) ? x : (__expf(x) - 1.0f);   // ELU
        _Float16 hv = (_Float16)y;
        int b = row >> 9, s = row & 511;
        long bh = (long)b * Hh + h;
        if      (p == 0) qh[(bh * Ss + s) * DHP + dh] = hv;
        else if (p == 1) kh[(bh * Ss + s) * DHP + dh] = hv;
        else             vT[(bh * DHP + dh) * Ss + s] = hv;
      }
    }
  }
}

// ---------------- flash attention: one wave per (b,h, 16-query tile) ----------------
__global__ __launch_bounds__(32)
void k_attn(const _Float16* __restrict__ qh, const _Float16* __restrict__ kh,
            const _Float16* __restrict__ vT, const int* __restrict__ num_docs,
            _Float16* __restrict__ ctx) {   // [MROWS][KP]
  __shared__ __align__(16) _Float16 Klds[32 * DHP];  // [key_local][dh]  4 KB
  __shared__ __align__(16) _Float16 Vlds[DHP * 32];  // [dh][key_local]  4 KB
  __shared__ __align__(16) _Float16 P[16 * 32];      // probs transpose  1 KB
  const int lane = threadIdx.x & 31;
  const int hi = lane >> 4, lo = lane & 15;
  const int qt = blockIdx.x & 31;
  const int h  = (blockIdx.x >> 5) & 7;
  const int b  = blockIdx.x >> 8;
  const int q0 = qt * 16;
  const int nd = num_docs[b];
  const long bh = (long)b * Hh + h;
  const _Float16* qb = qh + bh * Ss * DHP;
  const _Float16* kb = kh + bh * Ss * DHP;
  const _Float16* vb = vT + bh * DHP * Ss;

  // Q A-fragments covering dh = 0..63 (two K-steps of 32)
  FragAB qa[2];
  {
    const _Float16* qp = qb + (long)(q0 + lo) * DHP;
    qa[0].u[0] = ldg4(qp + hi * 8);
    qa[0].u[1] = ldg4(qp + 16 + hi * 8);
    qa[1].u[0] = ldg4(qp + 32 + hi * 8);
    qa[1].u[1] = ldg4(qp + 48 + hi * 8);
  }

  v8f o[4];
#pragma unroll
  for (int t = 0; t < 4; ++t) o[t] = v8f{};
  float mrow[8], lrow[8];
#pragma unroll
  for (int r = 0; r < 8; ++r) { mrow[r] = -3.0e38f; lrow[r] = 0.0f; }
  const float scale = 0.14142135623730950488f;  // 1/sqrt(50)

  for (int kt = 0; kt <= qt; kt += 2) {
    const int kb0 = kt * 16;                    // 32 keys this iteration

    // ---- stage K tile (contiguous 4KB) and V tile (64 rows x 64B) into LDS ----
    {
      const char* gk = (const char*)(kb + (long)kb0 * DHP);
      char* lk = (char*)Klds;
#pragma unroll
      for (int c = 0; c < 8; ++c) {
        int off = c * 512 + lane * 16;
        copy16_g2l(gk + off, lk + off);
      }
      const char* gv = (const char*)vb + (long)kb0 * 2;
      char* lv = (char*)Vlds;
      int vrow = lane >> 2;
      int vcol = (lane & 3) * 16;
#pragma unroll
      for (int c = 0; c < 8; ++c) {
        int row = c * 8 + vrow;
        copy16_g2l(gv + (long)row * (Ss * 2) + vcol, lv + row * 64 + vcol);
      }
      copy_wait();
    }

    // ---- scores: S = Q K^T from LDS K tile ----
    v8f sc[2]; sc[0] = v8f{}; sc[1] = v8f{};
#pragma unroll
    for (int half = 0; half < 2; ++half) {
      const char* kp = (const char*)Klds + (half * 16 + lo) * 128 + hi * 16;
      FragAB b0, b1;
      b0.u[0] = *(const v4u*)kp;         b0.u[1] = *(const v4u*)(kp + 32);
      b1.u[0] = *(const v4u*)(kp + 64);  b1.u[1] = *(const v4u*)(kp + 96);
      sc[half] = wmma16(qa[0], b0, sc[half]);
      sc[half] = wmma16(qa[1], b1, sc[half]);
    }

    // ---- scale + causal/doc masks, online max/sum, P to LDS ----
    float alpha[8];
#pragma unroll
    for (int r = 0; r < 8; ++r) {
      int qg = q0 + hi * 8 + r;
      int kg0 = kb0 + lo, kg1 = kb0 + 16 + lo;
      float v0 = sc[0][r] * scale;
      if (kg0 > qg) v0 -= NEG;
      if (kg0 >= nd) v0 -= NEG;
      float v1 = sc[1][r] * scale;
      if (kg1 > qg) v1 -= NEG;
      if (kg1 >= nd) v1 -= NEG;
      float mx = fmaxf(v0, v1);
#pragma unroll
      for (int d = 8; d >= 1; d >>= 1) mx = fmaxf(mx, __shfl_xor(mx, d, 32));
      float nm = fmaxf(mrow[r], mx);
      alpha[r] = __expf(mrow[r] - nm);
      mrow[r] = nm;
      float p0 = __expf(v0 - nm);
      float p1 = __expf(v1 - nm);
      float rs = p0 + p1;
#pragma unroll
      for (int d = 8; d >= 1; d >>= 1) rs += __shfl_xor(rs, d, 32);
      lrow[r] = lrow[r] * alpha[r] + rs;
      int row = hi * 8 + r;
      P[row * 32 + lo]      = (_Float16)p0;
      P[row * 32 + 16 + lo] = (_Float16)p1;
    }
#pragma unroll
    for (int t = 0; t < 4; ++t)
#pragma unroll
      for (int e = 0; e < 8; ++e) o[t][e] *= alpha[e];

    __syncthreads();
    // ---- PV: P as A-fragment from LDS, V B-fragments from LDS ----
    FragAB pa;
    const _Float16* pp = &P[lo * 32 + hi * 16];
    pa.u[0] = *(const v4u*)pp;
    pa.u[1] = *(const v4u*)(pp + 8);
#pragma unroll
    for (int t = 0; t < 4; ++t) {
      FragAB vf;
      const char* vp = (const char*)Vlds + (t * 16 + lo) * 64 + hi * 32;
      vf.u[0] = *(const v4u*)vp;
      vf.u[1] = *(const v4u*)(vp + 16);
      o[t] = wmma16(pa, vf, o[t]);
    }
    __syncthreads();
  }

  // normalize, query mask, store ctx as f16 [row][h*50+dh]
#pragma unroll
  for (int t = 0; t < 4; ++t) {
    int dh = t * 16 + lo;
    if (dh >= DH) continue;
#pragma unroll
    for (int r = 0; r < 8; ++r) {
      int qg = q0 + hi * 8 + r;
      float val = (qg < nd) ? (o[t][r] / lrow[r]) : 0.0f;
      long row = (long)b * Ss + qg;
      ctx[row * KP + h * DH + dh] = (_Float16)val;
    }
  }
}

// ---------------- classifier GEMM + scatter into 4 output segments ----------------
__global__ __launch_bounds__(32)
void k_cls(const _Float16* __restrict__ ctx, const _Float16* __restrict__ Wcat,
           const float* __restrict__ bcat, float* __restrict__ out) {
  const int lane = threadIdx.x & 31;
  const int hi = lane >> 4, lo = lane & 15;
  const int mt = blockIdx.x / NCHUNK_CLS;
  const int nc = blockIdx.x % NCHUNK_CLS;
  const int m0 = mt * 32, n0 = nc * 64;

  v8f acc[8];
#pragma unroll
  for (int t = 0; t < 8; ++t) acc[t] = v8f{};

  const _Float16* arow0 = ctx + (long)(m0 + lo) * KP;
  const _Float16* arow1 = ctx + (long)(m0 + 16 + lo) * KP;
  for (int k0 = 0; k0 < KP; k0 += 32) {
    FragAB a0, a1;
    a0.u[0] = ldg4(arow0 + k0 + hi * 8);
    a0.u[1] = ldg4(arow0 + k0 + 16 + hi * 8);
    a1.u[0] = ldg4(arow1 + k0 + hi * 8);
    a1.u[1] = ldg4(arow1 + k0 + 16 + hi * 8);
    if (k0 + 32 < KP) {
      __builtin_prefetch(arow0 + k0 + 32 + hi * 8, 0, 0);
      __builtin_prefetch(arow1 + k0 + 32 + hi * 8, 0, 0);
    }
#pragma unroll
    for (int t = 0; t < 4; ++t) {
      FragAB bf;
      const _Float16* bp = Wcat + (long)(n0 + t * 16 + lo) * KP + k0 + hi * 8;
      bf.u[0] = ldg4(bp);
      bf.u[1] = ldg4(bp + 16);
      acc[t]     = wmma16(a0, bf, acc[t]);
      acc[4 + t] = wmma16(a1, bf, acc[4 + t]);
    }
  }

#pragma unroll
  for (int t = 0; t < 4; ++t) {
    int n = n0 + t * 16 + lo;
    if (n >= NCLS) continue;
    float bval = bcat[n];
    long base; int width, off;
    if      (n < 25)  { base = 0;                 width = 25;  off = n;       }
    else if (n < 29)  { base = (long)MROWS * 25;  width = 4;   off = n - 25;  }
    else if (n < 591) { base = (long)MROWS * 29;  width = 562; off = n - 29;  }
    else              { base = (long)MROWS * 591; width = 7;   off = n - 591; }
#pragma unroll
    for (int half = 0; half < 2; ++half) {
#pragma unroll
      for (int r = 0; r < 8; ++r) {
        long row = m0 + half * 16 + hi * 8 + r;
        out[base + row * width + off] = acc[half * 4 + t][r] + bval;
      }
    }
  }
}

// ---------------- launch ----------------
extern "C" void kernel_launch(void* const* d_in, const int* in_sizes, int n_in,
                              void* d_out, int out_size, void* d_ws, size_t ws_size,
                              hipStream_t stream) {
  const float* X   = (const float*)d_in[0];
  const int*   nd  = (const int*)  d_in[1];
  const float* Wq  = (const float*)d_in[2];
  const float* bq  = (const float*)d_in[3];
  const float* Wk  = (const float*)d_in[4];
  const float* bk  = (const float*)d_in[5];
  const float* Wv  = (const float*)d_in[6];
  const float* bv  = (const float*)d_in[7];
  const float* Wc0 = (const float*)d_in[8];
  const float* bc0 = (const float*)d_in[9];
  const float* Wc1 = (const float*)d_in[10];
  const float* bc1 = (const float*)d_in[11];
  const float* Wc2 = (const float*)d_in[12];
  const float* bc2 = (const float*)d_in[13];
  const float* Wc3 = (const float*)d_in[14];
  const float* bc3 = (const float*)d_in[15];
  float* out = (float*)d_out;

  // workspace carve-up (256B aligned)
  char* ws = (char*)d_ws;
  size_t off = 0;
  auto carve = [&](size_t bytes) -> char* {
    char* p = ws + off;
    off = (off + bytes + 255) & ~(size_t)255;
    return p;
  };
  _Float16* Xh   = (_Float16*)carve((size_t)MROWS * KP * 2);
  _Float16* Wh   = (_Float16*)carve((size_t)3 * NQKV * KP * 2);
  _Float16* Wcat = (_Float16*)carve((size_t)NPAD * KP * 2);
  float*    bcat = (float*)   carve((size_t)NPAD * 4);
  size_t qkv_bytes = (size_t)Bb * Hh * Ss * DHP * 2;
  _Float16* qh   = (_Float16*)carve(qkv_bytes);
  _Float16* kh   = (_Float16*)carve(qkv_bytes);
  _Float16* vT   = (_Float16*)carve(qkv_bytes);
  size_t ctx_bytes = (size_t)MROWS * KP * 2;
  _Float16* ctx  = (_Float16*)carve(ctx_bytes);

  // zero padded regions (dh padding in q/k, K padding in ctx)
  (void)hipMemsetAsync(qh,  0, qkv_bytes, stream);
  (void)hipMemsetAsync(kh,  0, qkv_bytes, stream);
  (void)hipMemsetAsync(vT,  0, qkv_bytes, stream);
  (void)hipMemsetAsync(ctx, 0, ctx_bytes, stream);

  {
    long n = (long)MROWS * KP;
    k_cvt_x<<<dim3((unsigned)((n + 255) / 256)), 256, 0, stream>>>(X, Xh);
  }
  {
    long n = (long)3 * NQKV * KP + (long)NPAD * KP + NPAD;
    k_cvt_w<<<dim3((unsigned)((n + 255) / 256)), 256, 0, stream>>>(
        Wq, Wk, Wv, Wc0, Wc1, Wc2, Wc3, bc0, bc1, bc2, bc3, Wh, Wcat, bcat);
  }
  {
    dim3 grid((MROWS / 32) * NCHUNK_QKV, 3);
    k_qkv<<<grid, 32, 0, stream>>>(Xh, Wh, bq, bk, bv, qh, kh, vT);
  }
  {
    dim3 grid(Bb * Hh * (Ss / 16));
    k_attn<<<grid, 32, 0, stream>>>(qh, kh, vT, nd, ctx);
  }
  {
    dim3 grid((MROWS / 32) * NCHUNK_CLS);
    k_cls<<<grid, 32, 0, stream>>>(ctx, Wcat, bcat, out);
  }
}
